// RecSysTransformer_82471962018567
// MI455X (gfx1250) — matrix operand order repaired and checked by reading, not compiled, not run on hardware
//
#include <hip/hip_runtime.h>
#include <hip/hip_bf16.h>
#include <math.h>

// ---------------- model constants ----------------
#define NL   4
#define BBZ  2
#define TT   1024
#define DD   2048
#define HQn  16
#define HKVn 4
#define DKn  128
#define HIDn 5464

typedef __attribute__((ext_vector_type(16))) __bf16 v16bf;
typedef __attribute__((ext_vector_type(8)))  float  v8f;

union BfPack  { v16bf v; uint4 q[2]; };
union BfPack8 { __bf16 h[16]; uint4 q[2]; };

// =====================================================================
// Generic GEMM:  C[M,N] = (addC ? C : 0) + A[M,K] @ B[K,N]
// fp32 HBM -> bf16 LDS -> v_wmma_f32_16x16x32_bf16 (f32 accum).
// Block tile 128x128, K-step 32, 256 thr = 8 waves, wave = 64x32.
// Software-pipelined staging: tile k+1 loaded to regs while tile k computes.
// =====================================================================
#define BM 128
#define BN 128
#define BK 32
#define LSD 40   // padded LDS stride (bf16), rows stay 16B aligned (80B)

__global__ __launch_bounds__(256)
void gemm_wmma_bf16(const float* __restrict__ A, const float* __restrict__ Bw,
                    float* __restrict__ C, int M, int N, int K, int addC)
{
    __shared__ __align__(16) __bf16 As[BM][LSD];
    __shared__ __align__(16) __bf16 Bs[BN][LSD];   // transposed: Bs[n][k]

    const int tid  = threadIdx.x;
    const int lane = tid & 31;
    const int w    = tid >> 5;
    const int wm   = w >> 2;        // 0..1
    const int wn   = w & 3;         // 0..3
    const int ml   = lane & 15;
    const int kh   = lane >> 4;

    const int m0 = blockIdx.y * BM;
    const int n0 = blockIdx.x * BN;

    // staging ownership
    const int arow = tid >> 1;          // 0..127
    const int aseg = (tid & 1) * 16;    // col offset 0/16
    const int bn   = tid & 127;         // 0..127 (n within tile)
    const int bseg = (tid >> 7) * 16;   // k offset 0/16

    const bool fullN = (n0 + BN <= N);  // uniform

    v8f acc[4][2];
    #pragma unroll
    for (int i = 0; i < 4; ++i)
        #pragma unroll
        for (int j = 0; j < 2; ++j) acc[i][j] = {};

    float raf[16], rbf[16];

    // ---- register-tile loader (uniform fast/slow split) ----
    auto load_regs = [&](int k0) {
        const bool fullK = (k0 + BK <= K);   // uniform
        // A: 16 consecutive floats along K
        if (fullK) {
            const float* src = A + (size_t)(m0 + arow) * K + k0 + aseg;
            #pragma unroll
            for (int i = 0; i < 4; ++i) {
                float4 t = ((const float4*)src)[i];
                raf[4*i+0] = t.x; raf[4*i+1] = t.y; raf[4*i+2] = t.z; raf[4*i+3] = t.w;
            }
        } else {
            #pragma unroll
            for (int i = 0; i < 16; ++i) {
                int gc = k0 + aseg + i;
                raf[i] = (gc < K) ? A[(size_t)(m0 + arow) * K + gc] : 0.f;
            }
        }
        // B: 16 floats along K at fixed n (lane-coalesced along n)
        if (fullK && fullN) {
            const float* src = Bw + (size_t)(k0 + bseg) * N + n0 + bn;
            #pragma unroll
            for (int j = 0; j < 16; ++j) rbf[j] = src[(size_t)j * N];
        } else {
            int gn = n0 + bn;
            #pragma unroll
            for (int j = 0; j < 16; ++j) {
                int gk = k0 + bseg + j;
                rbf[j] = (gk < K && gn < N) ? Bw[(size_t)gk * N + gn] : 0.f;
            }
        }
    };

    auto store_lds = [&]() {
        BfPack8 pa;
        #pragma unroll
        for (int i = 0; i < 16; ++i) pa.h[i] = (__bf16)raf[i];
        uint4* da = (uint4*)&As[arow][aseg];
        da[0] = pa.q[0]; da[1] = pa.q[1];

        BfPack8 pb;
        #pragma unroll
        for (int i = 0; i < 16; ++i) pb.h[i] = (__bf16)rbf[i];
        uint4* db = (uint4*)&Bs[bn][bseg];
        db[0] = pb.q[0]; db[1] = pb.q[1];
    };

    const int ksteps = (K + BK - 1) / BK;
    load_regs(0);

    for (int kt = 0; kt < ksteps; ++kt) {
        store_lds();
        __syncthreads();

        if (kt + 1 < ksteps) {
            load_regs((kt + 1) * BK);                 // hide global latency
            if (kt + 2 < ksteps) {                    // L2 prefetch distance 2
                int pk = (kt + 2) * BK;
                __builtin_prefetch(&A[(size_t)(m0 + arow) * K + pk + aseg], 0, 0);
                if (fullN)
                    __builtin_prefetch(&Bw[(size_t)(pk + bseg) * N + n0 + bn], 0, 0);
            }
        }

        // ---- fragments per ISA layouts ----
        v16bf af[4], bfr[2];
        #pragma unroll
        for (int mf = 0; mf < 4; ++mf) {
            int rm = wm * 64 + mf * 16 + ml;
            BfPack p;
            p.q[0] = *(const uint4*)&As[rm][8 * kh];        // K = 8*kh + j
            p.q[1] = *(const uint4*)&As[rm][16 + 8 * kh];   // K = 16+8*kh+(j-8)
            af[mf] = p.v;
        }
        #pragma unroll
        for (int nf = 0; nf < 2; ++nf) {
            int cn = wn * 32 + nf * 16 + ml;
            BfPack p;
            p.q[0] = *(const uint4*)&Bs[cn][8 * kh];        // K = 8*kh + j
            p.q[1] = *(const uint4*)&Bs[cn][16 + 8 * kh];
            bfr[nf] = p.v;
        }
        #pragma unroll
        for (int mf = 0; mf < 4; ++mf)
            #pragma unroll
            for (int nf = 0; nf < 2; ++nf)
                acc[mf][nf] = __builtin_amdgcn_wmma_f32_16x16x32_bf16(
                    false, af[mf], false, bfr[nf], (short)0, acc[mf][nf], false, false);
        __syncthreads();
    }

    // ---- epilogue: C layout M = r + 8*kh, N = ml ----
    #pragma unroll
    for (int mf = 0; mf < 4; ++mf)
        #pragma unroll
        for (int nf = 0; nf < 2; ++nf)
            #pragma unroll
            for (int r = 0; r < 8; ++r) {
                int row = m0 + wm * 64 + mf * 16 + r + 8 * kh;
                int col = n0 + wn * 32 + nf * 16 + ml;
                if (row < M && col < N) {
                    size_t idx = (size_t)row * N + col;
                    float vv = acc[mf][nf][r];
                    if (addC) vv += C[idx];
                    C[idx] = vv;
                }
            }
}

// =====================================================================
// Attention: one wave per (b, head, 16-query tile).
// Logits row-block [16 x T] in LDS; K/V pulled from L2 (4 MB/layer each,
// resident in the 192 MB L2).
// =====================================================================
__global__ __launch_bounds__(32)
void attn_wmma(const float* __restrict__ q, const float* __restrict__ kk,
               const float* __restrict__ vv, float* __restrict__ o,
               const int* __restrict__ candp, int T)
{
    extern __shared__ float sm[];
    float* P    = sm;            // [16][T]
    float* rsum = sm + 16 * T;   // [16]

    const int lane = threadIdx.x;
    const int ml = lane & 15, kh = lane >> 4;
    const int qt = blockIdx.x, h = blockIdx.y, b = blockIdx.z;
    const int kvh  = h / (HQn / HKVn);
    const int cand = *candp;

    // ---- Q tile -> 4 A-fragments, float4 loads ----
    v16bf qa[4];
    {
        const int tq = qt * 16 + ml;
        const float* qr = q + ((size_t)(b * T + tq) * HQn + h) * DKn;
        #pragma unroll
        for (int kc = 0; kc < 4; ++kc) {
            const float4* p0 = (const float4*)(qr + kc * 32 + 16 * kh);
            const float4* p1 = (const float4*)(qr + kc * 32 + 16 * kh + 8);
            float4 u0 = p0[0], u1 = p0[1], v0 = p1[0], v1 = p1[1];
            v16bf a;
            a[0]=(__bf16)u0.x;  a[1]=(__bf16)u0.y;  a[2]=(__bf16)u0.z;  a[3]=(__bf16)u0.w;
            a[4]=(__bf16)u1.x;  a[5]=(__bf16)u1.y;  a[6]=(__bf16)u1.z;  a[7]=(__bf16)u1.w;
            a[8]=(__bf16)v0.x;  a[9]=(__bf16)v0.y;  a[10]=(__bf16)v0.z; a[11]=(__bf16)v0.w;
            a[12]=(__bf16)v1.x; a[13]=(__bf16)v1.y; a[14]=(__bf16)v1.z; a[15]=(__bf16)v1.w;
            qa[kc] = a;
        }
    }

    const int slen = qt * 16 + 16;

    // ---- logits: Q @ K^T, masked into LDS ----
    for (int s0 = 0; s0 < slen; s0 += 16) {
        const int key = s0 + ml;
        const float* kr = kk + ((size_t)(b * T + key) * HKVn + kvh) * DKn;
        v8f c = {};
        #pragma unroll
        for (int kc = 0; kc < 4; ++kc) {
            const float4* kp = (const float4*)(kr + kc * 32 + 16 * kh);
            float4 t0 = kp[0], t1 = kp[1], t2 = kp[2], t3 = kp[3];
            v16bf bfr;
            bfr[0]=(__bf16)t0.x;  bfr[1]=(__bf16)t0.y;  bfr[2]=(__bf16)t0.z;  bfr[3]=(__bf16)t0.w;
            bfr[4]=(__bf16)t1.x;  bfr[5]=(__bf16)t1.y;  bfr[6]=(__bf16)t1.z;  bfr[7]=(__bf16)t1.w;
            bfr[8]=(__bf16)t2.x;  bfr[9]=(__bf16)t2.y;  bfr[10]=(__bf16)t2.z; bfr[11]=(__bf16)t2.w;
            bfr[12]=(__bf16)t3.x; bfr[13]=(__bf16)t3.y; bfr[14]=(__bf16)t3.z; bfr[15]=(__bf16)t3.w;
            c = __builtin_amdgcn_wmma_f32_16x16x32_bf16(
                    false, qa[kc], false, bfr, (short)0, c, false, false);
        }
        #pragma unroll
        for (int r = 0; r < 8; ++r) {
            int row = r + 8 * kh;
            int tr  = qt * 16 + row;
            bool ok = (key <= tr) && !((tr >= cand) && (key >= cand) && (tr != key));
            P[row * T + key] = ok ? c[r] : -1e30f;
        }
    }

    // ---- softmax (2 lanes per row) ----
    {
        int row = lane >> 1;
        int c0  = lane & 1;
        float mx = -1e30f;
        for (int cc = c0; cc < slen; cc += 2) mx = fmaxf(mx, P[row * T + cc]);
        mx = fmaxf(mx, __shfl_xor(mx, 1, 32));
        float s = 0.f;
        for (int cc = c0; cc < slen; cc += 2) {
            float e = __expf(P[row * T + cc] - mx);
            P[row * T + cc] = e;
            s += e;
        }
        s += __shfl_xor(s, 1, 32);
        if (c0 == 0) rsum[row] = s;
    }

    // ---- O = P @ V ----
    v8f acc[8];
    #pragma unroll
    for (int i = 0; i < 8; ++i) acc[i] = {};
    for (int s0 = 0; s0 < slen; s0 += 32) {
        // P in A-layout: two runs of 8 consecutive keys, run-uniform guards
        v16bf pa;
        {
            int base0 = s0 + 8 * kh;         // j=0..7
            int base1 = s0 + 16 + 8 * kh;    // j=8..15
            float f0[8] = {}, f1[8] = {};
            if (base0 < slen) {
                const float4* pp = (const float4*)&P[ml * T + base0];
                float4 a0 = pp[0], a1 = pp[1];
                f0[0]=a0.x; f0[1]=a0.y; f0[2]=a0.z; f0[3]=a0.w;
                f0[4]=a1.x; f0[5]=a1.y; f0[6]=a1.z; f0[7]=a1.w;
            }
            if (base1 < slen) {
                const float4* pp = (const float4*)&P[ml * T + base1];
                float4 a0 = pp[0], a1 = pp[1];
                f1[0]=a0.x; f1[1]=a0.y; f1[2]=a0.z; f1[3]=a0.w;
                f1[4]=a1.x; f1[5]=a1.y; f1[6]=a1.z; f1[7]=a1.w;
            }
            #pragma unroll
            for (int j = 0; j < 8; ++j) { pa[j] = (__bf16)f0[j]; pa[8+j] = (__bf16)f1[j]; }
        }
        #pragma unroll
        for (int nf = 0; nf < 8; ++nf) {
            v16bf vb;
            int dk = nf * 16 + ml;
            #pragma unroll
            for (int j = 0; j < 16; ++j) {              // strided gather, lane-coalesced
                int key = s0 + 8 * kh + j + (j >= 8 ? 8 : 0);
                float x = (key < slen)
                          ? vv[((size_t)(b * T + key) * HKVn + kvh) * DKn + dk] : 0.f;
                vb[j] = (__bf16)x;
            }
            acc[nf] = __builtin_amdgcn_wmma_f32_16x16x32_bf16(
                          false, pa, false, vb, (short)0, acc[nf], false, false);
        }
    }

    #pragma unroll
    for (int nf = 0; nf < 8; ++nf)
        #pragma unroll
        for (int r = 0; r < 8; ++r) {
            int row = r + 8 * kh;
            int t   = qt * 16 + row;
            float inv = 1.0f / rsum[row];
            o[((size_t)(b * T + t) * HQn + h) * DKn + nf * 16 + ml] = acc[nf][r] * inv;
        }
}

// =====================================================================
// RMSNorm (one block per token row). In-place safe.
// =====================================================================
__global__ __launch_bounds__(256)
void rmsnorm_kernel(const float* __restrict__ x, const float* __restrict__ g,
                    float* __restrict__ y, int D)
{
    __shared__ float red[8];
    __shared__ float rfac;
    const int row = blockIdx.x;
    const int tid = threadIdx.x;
    const float* xr = x + (size_t)row * D;

    float s = 0.f;
    for (int i = tid; i < D; i += 256) { float v = xr[i]; s += v * v; }
    #pragma unroll
    for (int off = 16; off > 0; off >>= 1) s += __shfl_xor(s, off, 32);
    if ((tid & 31) == 0) red[tid >> 5] = s;
    __syncthreads();
    if (tid == 0) {
        float t = 0.f;
        #pragma unroll
        for (int i = 0; i < 8; ++i) t += red[i];
        rfac = rsqrtf(t / (float)D + 1e-5f);
    }
    __syncthreads();
    float r = rfac;
    for (int i = tid; i < D; i += 256) y[(size_t)row * D + i] = g[i] * xr[i] * r;
}

// =====================================================================
// RoPE (in place). One thread per (b,t,head,pair).
// =====================================================================
__global__ void rope_kernel(float* __restrict__ x, int T, int H)
{
    int idx = blockIdx.x * blockDim.x + threadIdx.x;
    int total = BBZ * T * H * 64;
    if (idx >= total) return;
    int p  = idx & 63;
    int hh = (idx >> 6) % H;
    int t  = (idx / (64 * H)) % T;
    int b  = idx / (64 * H * T);
    float inv = __powf(10000.0f, -(float)(2 * p) / 128.0f);
    float sn, cs;
    __sincosf((float)t * inv, &sn, &cs);
    size_t base = ((size_t)(b * T + t) * H + hh) * DKn;
    float x1 = x[base + p], x2 = x[base + 64 + p];
    x[base + p]      = x1 * cs - x2 * sn;
    x[base + 64 + p] = x2 * cs + x1 * sn;
}

// =====================================================================
// g <- gelu_exact(g) * u
// =====================================================================
__global__ void gelu_mul_kernel(float* __restrict__ g, const float* __restrict__ u,
                                size_t n)
{
    size_t i = (size_t)blockIdx.x * blockDim.x + threadIdx.x;
    if (i >= n) return;
    float x = g[i];
    float ge = 0.5f * x * (1.0f + erff(x * 0.70710678118654752f));
    g[i] = ge * u[i];
}

// =====================================================================
// launch
// =====================================================================
extern "C" void kernel_launch(void* const* d_in, const int* in_sizes, int n_in,
                              void* d_out, int out_size, void* d_ws, size_t ws_size,
                              hipStream_t stream)
{
    (void)in_sizes; (void)n_in; (void)out_size; (void)ws_size;

    const float* x   = (const float*)d_in[0];
    const float* s1  = (const float*)d_in[1];
    const float* s2  = (const float*)d_in[2];
    const float* s3  = (const float*)d_in[3];
    const float* wq  = (const float*)d_in[4];
    const float* wk  = (const float*)d_in[5];
    const float* wv  = (const float*)d_in[6];
    const float* wo  = (const float*)d_in[7];
    const float* wg  = (const float*)d_in[8];
    const float* wvl = (const float*)d_in[9];
    const float* wou = (const float*)d_in[10];
    const int*   cand = (const int*)d_in[11];

    const size_t M = (size_t)BBZ * TT;     // 2048 token rows
    float* h = (float*)d_out;

    float* ws = (float*)d_ws;
    float* hn = ws;  ws += M * DD;
    float* qb = ws;  ws += M * (HQn  * DKn);
    float* kb = ws;  ws += M * (HKVn * DKn);
    float* vb = ws;  ws += M * (HKVn * DKn);
    float* ob = ws;  ws += M * (HQn  * DKn);
    float* gb = ws;  ws += M * HIDn;
    float* ub = ws;  ws += M * HIDn;

    hipMemcpyAsync(h, x, M * DD * sizeof(float), hipMemcpyDeviceToDevice, stream);

    const unsigned Mb = (unsigned)(M / 128);
    const size_t attn_smem = ((size_t)16 * TT + 16) * sizeof(float);

    for (int l = 0; l < NL; ++l) {
        rmsnorm_kernel<<<(unsigned)M, 256, 0, stream>>>(h, s1 + (size_t)l * DD, hn, DD);

        gemm_wmma_bf16<<<dim3((HQn * DKn) / 128, Mb), 256, 0, stream>>>(
            hn, wq + (size_t)l * DD * HQn * DKn, qb, (int)M, HQn * DKn, DD, 0);
        gemm_wmma_bf16<<<dim3((HKVn * DKn) / 128, Mb), 256, 0, stream>>>(
            hn, wk + (size_t)l * DD * HKVn * DKn, kb, (int)M, HKVn * DKn, DD, 0);
        gemm_wmma_bf16<<<dim3((HKVn * DKn) / 128, Mb), 256, 0, stream>>>(
            hn, wv + (size_t)l * DD * HKVn * DKn, vb, (int)M, HKVn * DKn, DD, 0);

        {
            int tot = BBZ * TT * HQn * 64;
            rope_kernel<<<(tot + 255) / 256, 256, 0, stream>>>(qb, TT, HQn);
            tot = BBZ * TT * HKVn * 64;
            rope_kernel<<<(tot + 255) / 256, 256, 0, stream>>>(kb, TT, HKVn);
        }

        attn_wmma<<<dim3(TT / 16, HQn, BBZ), 32, attn_smem, stream>>>(
            qb, kb, vb, ob, cand, TT);

        gemm_wmma_bf16<<<dim3(DD / 128, Mb), 256, 0, stream>>>(
            ob, wo + (size_t)l * HQn * DKn * DD, h, (int)M, DD, HQn * DKn, 1);

        rmsnorm_kernel<<<(unsigned)M, 256, 0, stream>>>(h, s2 + (size_t)l * DD, hn, DD);

        gemm_wmma_bf16<<<dim3((HIDn + 127) / 128, Mb), 256, 0, stream>>>(
            hn, wg + (size_t)l * DD * HIDn, gb, (int)M, HIDn, DD, 0);
        gemm_wmma_bf16<<<dim3((HIDn + 127) / 128, Mb), 256, 0, stream>>>(
            hn, wvl + (size_t)l * DD * HIDn, ub, (int)M, HIDn, DD, 0);

        {
            size_t n = M * (size_t)HIDn;
            gelu_mul_kernel<<<(unsigned)((n + 255) / 256), 256, 0, stream>>>(gb, ub, n);
        }

        gemm_wmma_bf16<<<dim3(DD / 128, Mb), 256, 0, stream>>>(
            gb, wou + (size_t)l * HIDn * DD, h, (int)M, DD, HIDn, 1);

        rmsnorm_kernel<<<(unsigned)M, 256, 0, stream>>>(h, s3 + (size_t)l * DD, h, DD);
    }
}
